// CausalScaledDotAttention_19447611916535
// MI455X (gfx1250) — matrix operand order repaired and checked by reading, not compiled, not run on hardware
//
#include <hip/hip_runtime.h>
#include <hip/hip_bf16.h>

typedef __bf16 v16bf __attribute__((ext_vector_type(16)));
typedef __bf16 v8bf  __attribute__((ext_vector_type(8)));
typedef float  v8f   __attribute__((ext_vector_type(8)));

static constexpr int Hd = 1024;   // hidden dim
static constexpr int Sd = 2048;   // sequence length (Q == S)
static constexpr int Bd = 16;     // batch
#define NEG_BIG (-10000000.0f)

// ---------------------------------------------------------------- helpers

__device__ __forceinline__ v8f vzero8() {
  v8f z;
#pragma unroll
  for (int i = 0; i < 8; ++i) z[i] = 0.0f;
  return z;
}

__device__ __forceinline__ v8f wmma_bf16(v16bf a, v16bf b, v8f c) {
  // D = A(16x32 bf16) * B(32x16 bf16) + C(16x16 f32)
  return __builtin_amdgcn_wmma_f32_16x16x32_bf16(
      /*neg_a=*/false, a, /*neg_b=*/false, b,
      /*c_mod=*/(short)0, c, /*reuse_a=*/false, /*reuse_b=*/false);
}

// A-fragment 16x32 (M x K), source row-major contiguous in K.
// CDNA5 layout: lanes 0-15 -> M=lane, K 0-7 (v0-3) & 16-23 (v4-7);
//               lanes 16-31 -> M=lane-16, K 8-15 & 24-31.
__device__ __forceinline__ v16bf frag_a_bf16(const __bf16* base, int lda, int lane) {
  const int l16 = lane & 15, half = lane >> 4;
  const __bf16* p = base + (size_t)l16 * lda + half * 8;
  union { v16bf v; v8bf h[2]; } u;
  u.h[0] = *(const v8bf*)(p);
  u.h[1] = *(const v8bf*)(p + 16);
  return u.v;
}

// Same A layout but converting f32 source on the fly.
__device__ __forceinline__ v16bf frag_a_f32(const float* base, int lda, int lane) {
  const int l16 = lane & 15, half = lane >> 4;
  const float* p = base + (size_t)l16 * lda + half * 8;
  v16bf r;
#pragma unroll
  for (int i = 0; i < 8; ++i) {
    r[i]     = (__bf16)p[i];
    r[i + 8] = (__bf16)p[i + 16];
  }
  return r;
}

// B-fragment 32x16 (K x N). Storage is [N][K]: each output column n is a
// contiguous K-row (ldb elements apart). Lanes 0-15 -> N=lane, K 0-7 & 16-23;
// lanes 16-31 -> N=lane-16, K 8-15 & 24-31. Two 16-byte loads per lane.
__device__ __forceinline__ v16bf frag_b_bf16(const __bf16* base, int ldb, int lane) {
  const int l16 = lane & 15, half = lane >> 4;
  const __bf16* p = base + (size_t)l16 * ldb + half * 8;
  union { v16bf v; v8bf h[2]; } u;
  u.h[0] = *(const v8bf*)(p);
  u.h[1] = *(const v8bf*)(p + 16);
  return u.v;
}

// ---------------------------------------------------------------- kernels

__global__ __launch_bounds__(256) void cvt_f32_to_bf16(const float* __restrict__ src,
                                                       __bf16* __restrict__ dst, int n) {
  int i = blockIdx.x * 256 + threadIdx.x;
  if (i < n) dst[i] = (__bf16)src[i];
}

// Y = X @ W^T + b ; X f32 [M x Hd], W bf16 [Hd x Hd] row-major, Y bf16.
// transposeOut=1 stores per-batch transposed [B][Hd][Sd] (used for V).
// Ping-pong double-buffered K-loop: no register copies between stages.
__global__ __launch_bounds__(128) void qkv_proj_gemm(const float* __restrict__ X,
                                                     const __bf16* __restrict__ W,
                                                     const float* __restrict__ bias,
                                                     __bf16* __restrict__ Y,
                                                     int transposeOut) {
  const int lane = threadIdx.x & 31;
  const int wave = threadIdx.x >> 5;
  const int m0 = blockIdx.y * 64 + wave * 16;
  const int n0 = blockIdx.x * 64;

  v8f acc[4];
#pragma unroll
  for (int t = 0; t < 4; ++t) acc[t] = vzero8();

  const float* Abase = X + (size_t)m0 * Hd;

  v16bf a0, a1, b0[4], b1[4];
  a0 = frag_a_f32(Abase, Hd, lane);
#pragma unroll
  for (int t = 0; t < 4; ++t)
    b0[t] = frag_b_bf16(W + (size_t)(n0 + t * 16) * Hd, Hd, lane);

  for (int k0 = 0; k0 < Hd - 64; k0 += 64) {
    a1 = frag_a_f32(Abase + k0 + 32, Hd, lane);
#pragma unroll
    for (int t = 0; t < 4; ++t)
      b1[t] = frag_b_bf16(W + (size_t)(n0 + t * 16) * Hd + k0 + 32, Hd, lane);
#pragma unroll
    for (int t = 0; t < 4; ++t) acc[t] = wmma_bf16(a0, b0[t], acc[t]);

    a0 = frag_a_f32(Abase + k0 + 64, Hd, lane);
#pragma unroll
    for (int t = 0; t < 4; ++t)
      b0[t] = frag_b_bf16(W + (size_t)(n0 + t * 16) * Hd + k0 + 64, Hd, lane);
#pragma unroll
    for (int t = 0; t < 4; ++t) acc[t] = wmma_bf16(a1, b1[t], acc[t]);
  }
  // epilogue: K-steps at Hd-64 (in b0/a0) and Hd-32
  a1 = frag_a_f32(Abase + Hd - 32, Hd, lane);
#pragma unroll
  for (int t = 0; t < 4; ++t)
    b1[t] = frag_b_bf16(W + (size_t)(n0 + t * 16) * Hd + Hd - 32, Hd, lane);
#pragma unroll
  for (int t = 0; t < 4; ++t) acc[t] = wmma_bf16(a0, b0[t], acc[t]);
#pragma unroll
  for (int t = 0; t < 4; ++t) acc[t] = wmma_bf16(a1, b1[t], acc[t]);

  const int l16 = lane & 15, half = lane >> 4;
#pragma unroll
  for (int t = 0; t < 4; ++t) {
    const int n = n0 + t * 16 + l16;
    const float bv = bias[n];
#pragma unroll
    for (int r = 0; r < 8; ++r) {
      const int m = m0 + r + half * 8;
      const __bf16 o = (__bf16)(acc[t][r] + bv);
      if (!transposeOut) {
        Y[(size_t)m * Hd + n] = o;
      } else {
        const int bb = m >> 11;           // m / Sd
        const int mm = m & (Sd - 1);      // m % Sd
        Y[(size_t)bb * Hd * Sd + (size_t)n * Sd + mm] = o;
      }
    }
  }
}

// One block = one (batch, 16-query-row strip). Computes the full 16 x Sd
// score strip with WMMA into LDS, causal mask + softmax in LDS, then writes
// the transposed attention output [B][S][Q] (f32) and bf16 P to workspace.
// k-tiles interleaved across waves so the causal prefix is load-balanced.
__global__ __launch_bounds__(128) void attn_scores_softmax(const __bf16* __restrict__ Qm,
                                                           const __bf16* __restrict__ Km,
                                                           float* __restrict__ attnOut,
                                                           __bf16* __restrict__ P) {
  __shared__ float sc[16][Sd];            // 128 KB score strip

  const int b  = blockIdx.x >> 7;         // Sd/16 = 128 strips per batch
  const int qt = blockIdx.x & 127;
  const int q0 = qt * 16;
  const int tid  = threadIdx.x;
  const int lane = tid & 31;
  const int wave = tid >> 5;
  const int l16 = lane & 15, half = lane >> 4;
  const float scale = 0.03125f;           // 1/sqrt(1024)

  const __bf16* Abase = Qm + ((size_t)b * Sd + q0) * Hd;

  // group g: wave w owns tiles j = g*16 + w*4 + t (t = 0..3) -> interleaved,
  // so the active (unmasked) tile prefix is split evenly across waves.
  for (int g = 0; g < 8; ++g) {
    const int jbase = g * 16 + wave * 4;
    v8f acc[4];
#pragma unroll
    for (int t = 0; t < 4; ++t) acc[t] = vzero8();

    // wave-uniform: skip fully-masked 4-tile groups (EXEC stays all-ones)
    if (jbase * 16 <= q0 + 15) {
      const __bf16* Bbase = Km + ((size_t)b * Sd + (size_t)jbase * 16) * Hd;

      v16bf a0, a1, b0[4], b1[4];
      a0 = frag_a_bf16(Abase, Hd, lane);
#pragma unroll
      for (int t = 0; t < 4; ++t)
        b0[t] = frag_b_bf16(Bbase + (size_t)t * 16 * Hd, Hd, lane);

      for (int k0 = 0; k0 < Hd - 64; k0 += 64) {
        a1 = frag_a_bf16(Abase + k0 + 32, Hd, lane);
#pragma unroll
        for (int t = 0; t < 4; ++t)
          b1[t] = frag_b_bf16(Bbase + (size_t)t * 16 * Hd + k0 + 32, Hd, lane);
#pragma unroll
        for (int t = 0; t < 4; ++t) acc[t] = wmma_bf16(a0, b0[t], acc[t]);

        a0 = frag_a_bf16(Abase + k0 + 64, Hd, lane);
#pragma unroll
        for (int t = 0; t < 4; ++t)
          b0[t] = frag_b_bf16(Bbase + (size_t)t * 16 * Hd + k0 + 64, Hd, lane);
#pragma unroll
        for (int t = 0; t < 4; ++t) acc[t] = wmma_bf16(a1, b1[t], acc[t]);
      }
      a1 = frag_a_bf16(Abase + Hd - 32, Hd, lane);
#pragma unroll
      for (int t = 0; t < 4; ++t)
        b1[t] = frag_b_bf16(Bbase + (size_t)t * 16 * Hd + Hd - 32, Hd, lane);
#pragma unroll
      for (int t = 0; t < 4; ++t) acc[t] = wmma_bf16(a0, b0[t], acc[t]);
#pragma unroll
      for (int t = 0; t < 4; ++t) acc[t] = wmma_bf16(a1, b1[t], acc[t]);
    }

#pragma unroll
    for (int t = 0; t < 4; ++t) {
      const int scol = (jbase + t) * 16 + l16;
#pragma unroll
      for (int r = 0; r < 8; ++r) {
        const int qrow = r + half * 8;    // 0..15 within strip
        float v = acc[t][r] * scale;
        if (scol > q0 + qrow) v = NEG_BIG;
        sc[qrow][scol] = v;
      }
    }
  }
  __syncthreads();

  // row softmax: wave w handles rows w, w+4, w+8, w+12 (wave32 shuffles)
  for (int row = wave; row < 16; row += 4) {
    float m = NEG_BIG;
    for (int s = lane; s < Sd; s += 32) m = fmaxf(m, sc[row][s]);
#pragma unroll
    for (int off = 16; off; off >>= 1) m = fmaxf(m, __shfl_xor(m, off));
    float sum = 0.0f;
    for (int s = lane; s < Sd; s += 32) {
      const float e = __expf(sc[row][s] - m);
      sc[row][s] = e;
      sum += e;
    }
#pragma unroll
    for (int off = 16; off; off >>= 1) sum += __shfl_xor(sum, off);
    const float inv = 1.0f / sum;
    for (int s = lane; s < Sd; s += 32) sc[row][s] *= inv;
  }
  __syncthreads();

  // transposed attention output: attnOut[b][s][q0..q0+15], 64B per s
  {
    float* dst = attnOut + (size_t)b * Sd * Sd + q0;
    for (int s = tid; s < Sd; s += 128) {
      float tmp[16];
#pragma unroll
      for (int r = 0; r < 16; ++r) tmp[r] = sc[r][s];
      float4* o = (float4*)(dst + (size_t)s * Sd);
      o[0] = make_float4(tmp[0], tmp[1], tmp[2], tmp[3]);
      o[1] = make_float4(tmp[4], tmp[5], tmp[6], tmp[7]);
      o[2] = make_float4(tmp[8], tmp[9], tmp[10], tmp[11]);
      o[3] = make_float4(tmp[12], tmp[13], tmp[14], tmp[15]);
    }
  }

  // bf16 P row-major [b][q][s] for the context GEMM (A-matrix, contiguous K)
  {
    const int r  = tid >> 3;              // 0..15
    const int s0 = (tid & 7) * 2;
    __bf16* pd = P + ((size_t)b * Sd + q0 + r) * Sd;
    for (int s = s0; s < Sd; s += 16) {
      union { __bf16 h[2]; unsigned int u; } pk;
      pk.h[0] = (__bf16)sc[r][s];
      pk.h[1] = (__bf16)sc[r][s + 1];
      *(unsigned int*)(pd + s) = pk.u;
    }
  }
}

// context[b] = P[b] (Q x S, bf16) @ V[b] (S x H) with V stored transposed
// [b][h][s] so both operands are contiguous along K=s. Causal: P[q][s]=0 for
// s>q -> truncate the K loop per wave tile. Ping-pong pipelined K-loop with a
// runtime trip count (kend is a multiple of 32, >= 32).
__global__ __launch_bounds__(128) void context_gemm(const __bf16* __restrict__ P,
                                                    const __bf16* __restrict__ VT,
                                                    float* __restrict__ ctx) {
  const int lane = threadIdx.x & 31;
  const int wave = threadIdx.x >> 5;
  const int bb = blockIdx.z;
  const int m0 = blockIdx.y * 64 + wave * 16;   // q rows
  const int n0 = blockIdx.x * 64;               // h cols

  const __bf16* Bm = VT + (size_t)bb * Hd * Sd;

  v8f acc[4];
#pragma unroll
  for (int t = 0; t < 4; ++t) acc[t] = vzero8();

  int kend = ((m0 + 16 + 31) >> 5) << 5;        // causal truncation, mult of 32
  if (kend > Sd) kend = Sd;

  const __bf16* Abase = P + (size_t)bb * Sd * Sd + (size_t)m0 * Sd;

  v16bf a0, a1, b0[4], b1[4];
  a0 = frag_a_bf16(Abase, Sd, lane);
#pragma unroll
  for (int t = 0; t < 4; ++t)
    b0[t] = frag_b_bf16(Bm + (size_t)(n0 + t * 16) * Sd, Sd, lane);

  int k0 = 0;
  for (; k0 + 64 < kend; k0 += 64) {
    a1 = frag_a_bf16(Abase + k0 + 32, Sd, lane);
#pragma unroll
    for (int t = 0; t < 4; ++t)
      b1[t] = frag_b_bf16(Bm + (size_t)(n0 + t * 16) * Sd + k0 + 32, Sd, lane);
#pragma unroll
    for (int t = 0; t < 4; ++t) acc[t] = wmma_bf16(a0, b0[t], acc[t]);

    a0 = frag_a_bf16(Abase + k0 + 64, Sd, lane);
#pragma unroll
    for (int t = 0; t < 4; ++t)
      b0[t] = frag_b_bf16(Bm + (size_t)(n0 + t * 16) * Sd + k0 + 64, Sd, lane);
#pragma unroll
    for (int t = 0; t < 4; ++t) acc[t] = wmma_bf16(a1, b1[t], acc[t]);
  }
  if (k0 + 32 < kend) {                         // two K-steps remain
    a1 = frag_a_bf16(Abase + k0 + 32, Sd, lane);
#pragma unroll
    for (int t = 0; t < 4; ++t)
      b1[t] = frag_b_bf16(Bm + (size_t)(n0 + t * 16) * Sd + k0 + 32, Sd, lane);
#pragma unroll
    for (int t = 0; t < 4; ++t) acc[t] = wmma_bf16(a0, b0[t], acc[t]);
#pragma unroll
    for (int t = 0; t < 4; ++t) acc[t] = wmma_bf16(a1, b1[t], acc[t]);
  } else {                                      // one K-step remains
#pragma unroll
    for (int t = 0; t < 4; ++t) acc[t] = wmma_bf16(a0, b0[t], acc[t]);
  }

  const int l16 = lane & 15, half = lane >> 4;
#pragma unroll
  for (int t = 0; t < 4; ++t) {
    const int n = n0 + t * 16 + l16;
#pragma unroll
    for (int r = 0; r < 8; ++r) {
      const int m = m0 + r + half * 8;
      ctx[(size_t)bb * Sd * Hd + (size_t)m * Hd + n] = acc[t][r];
    }
  }
}

// ---------------------------------------------------------------- launch

extern "C" void kernel_launch(void* const* d_in, const int* in_sizes, int n_in,
                              void* d_out, int out_size, void* d_ws, size_t ws_size,
                              hipStream_t stream) {
  (void)in_sizes; (void)n_in; (void)out_size; (void)ws_size;

  const float* queries = (const float*)d_in[0];
  const float* keys    = (const float*)d_in[1];
  const float* values  = (const float*)d_in[2];
  const float* Wq = (const float*)d_in[3];
  const float* bq = (const float*)d_in[4];
  const float* Wk = (const float*)d_in[5];
  const float* bk = (const float*)d_in[6];
  const float* Wv = (const float*)d_in[7];
  const float* bv = (const float*)d_in[8];

  float* ctxOut  = (float*)d_out;                                   // [B,Q,H]
  float* attnOut = (float*)d_out + (size_t)Bd * Sd * Hd;            // [B,S,Q]

  // workspace carve-up (256B aligned)
  size_t off = 0;
  auto carve = [&](size_t bytes) {
    void* p = (char*)d_ws + off;
    off += (bytes + 255) & ~(size_t)255;
    return p;
  };
  __bf16* Wqb = (__bf16*)carve((size_t)Hd * Hd * 2);
  __bf16* Wkb = (__bf16*)carve((size_t)Hd * Hd * 2);
  __bf16* Wvb = (__bf16*)carve((size_t)Hd * Hd * 2);
  __bf16* Qb  = (__bf16*)carve((size_t)Bd * Sd * Hd * 2);
  __bf16* Kb  = (__bf16*)carve((size_t)Bd * Sd * Hd * 2);
  __bf16* VTb = (__bf16*)carve((size_t)Bd * Hd * Sd * 2);
  __bf16* Pb  = (__bf16*)carve((size_t)Bd * Sd * Sd * 2);

  // 1) weights f32 -> bf16
  const int wn = Hd * Hd;
  cvt_f32_to_bf16<<<wn / 256, 256, 0, stream>>>(Wq, Wqb, wn);
  cvt_f32_to_bf16<<<wn / 256, 256, 0, stream>>>(Wk, Wkb, wn);
  cvt_f32_to_bf16<<<wn / 256, 256, 0, stream>>>(Wv, Wvb, wn);

  // 2) QKV projections (M = B*S = 32768)
  dim3 gGemm(Hd / 64, (Bd * Sd) / 64);   // (16, 512)
  qkv_proj_gemm<<<gGemm, 128, 0, stream>>>(queries, Wqb, bq, Qb, 0);
  qkv_proj_gemm<<<gGemm, 128, 0, stream>>>(keys,    Wkb, bk, Kb, 0);
  qkv_proj_gemm<<<gGemm, 128, 0, stream>>>(values,  Wvb, bv, VTb, 1);

  // 3) scores + causal mask + softmax + transposed attn output + bf16 P
  attn_scores_softmax<<<Bd * (Sd / 16), 128, 0, stream>>>(Qb, Kb, attnOut, Pb);

  // 4) context = P @ V
  dim3 gCtx(Hd / 64, Sd / 64, Bd);       // (16, 32, 16)
  context_gemm<<<gCtx, 128, 0, stream>>>(Pb, VTb, ctxOut);
}
